// OneTower_59983513255984
// MI455X (gfx1250) — matrix operand order repaired and checked by reading, not compiled, not run on hardware
//
#include <hip/hip_runtime.h>
#include <hip/hip_bf16.h>

#define B_  32
#define L_  512
#define LH_ 2048
#define D_  1024
#define H_  4
#define DK_ 256
#define NEGV -1000000000.0f

typedef __attribute__((ext_vector_type(16))) __bf16 bf16x16;
typedef __attribute__((ext_vector_type(8)))  float  f32x8;

// ---------------------------------------------------------------------------
// Generic bf16 WMMA GEMM:  C[M x N] = A[M x K] * Bt[N x K]^T   (f32 accumulate)
// Workgroup tile 128x128, K-step 32, 8 waves (4x2), each wave 32x64 (2x4 WMMA).
// MODE 0: bf16 out (+opt bias)    MODE 1: bf16 out into V^T layout (+bias)
// MODE 2: f32 out * scale         MODE 3: f32 out + bias
// MODE 4: additive-attention score epilogue (tanh, dot wa, atomicAdd)
// ---------------------------------------------------------------------------
template<int MODE>
__global__ __launch_bounds__(256) void gemm_wmma(
    const __bf16* __restrict__ A, const __bf16* __restrict__ Bt, void* __restrict__ Cv,
    const float* __restrict__ bias, int Ksz, int lda, int ldb, int ldc,
    int zInner, long sAo, long sAi, long sBo, long sBi, long sCo, long sCi,
    float scale, const float* __restrict__ aqv, const float* __restrict__ wav,
    float* __restrict__ ascore)
{
    __shared__ __bf16 As[128][40];   // 80B row stride: 16B-aligned, conflict-padded
    __shared__ __bf16 Bs[128][40];

    const int z  = blockIdx.z;
    const int zo = z / zInner, zi = z - zo * zInner;
    A  += (size_t)zo * sAo + (size_t)zi * sAi;
    Bt += (size_t)zo * sBo + (size_t)zi * sBi;
    const long coff = zo * sCo + zi * sCi;

    const int tileM = blockIdx.y * 128;
    const int tileN = blockIdx.x * 128;

    const int tid  = threadIdx.x;
    const int lane = tid & 31;
    const int wid  = tid >> 5;
    const int wm   = (wid >> 1) * 32;   // wave M offset (4 waves in M)
    const int wn   = (wid & 1)  * 64;   // wave N offset (2 waves in N)
    const int half = lane >> 4;         // lane half (0/1)
    const int r16  = lane & 15;

    const int lr = tid >> 1;            // cooperative-load row (0..127)
    const int lc = (tid & 1) * 16;      // cooperative-load half-chunk (0/16 halves)

    f32x8 acc[2][4];
    const f32x8 zf = {0.f,0.f,0.f,0.f,0.f,0.f,0.f,0.f};
    #pragma unroll
    for (int i = 0; i < 2; i++)
        #pragma unroll
        for (int j = 0; j < 4; j++) acc[i][j] = zf;

    const __bf16* aRow = A  + (size_t)(tileM + lr) * lda + lc;
    const __bf16* bRow = Bt + (size_t)(tileN + lr) * ldb + lc;

    for (int kb = 0; kb < Ksz; kb += 32) {
        uint4 a0 = *(const uint4*)(aRow + kb);
        uint4 a1 = *(const uint4*)(aRow + kb + 8);
        uint4 b0 = *(const uint4*)(bRow + kb);
        uint4 b1 = *(const uint4*)(bRow + kb + 8);
        __syncthreads();
        *(uint4*)&As[lr][lc]     = a0;
        *(uint4*)&As[lr][lc + 8] = a1;
        *(uint4*)&Bs[lr][lc]     = b0;
        *(uint4*)&Bs[lr][lc + 8] = b1;
        __syncthreads();

        bf16x16 af[2], bfr[4];
        #pragma unroll
        for (int i = 0; i < 2; i++) {
            int row = wm + i * 16 + r16;
            // A frag: lane-half h holds K = {8h..8h+7, 16+8h..16+8h+7}
            ((uint4*)&af[i])[0] = *(const uint4*)&As[row][half * 8];
            ((uint4*)&af[i])[1] = *(const uint4*)&As[row][16 + half * 8];
        }
        #pragma unroll
        for (int j = 0; j < 4; j++) {
            int n = wn + j * 16 + r16;
            // B frag: lane-half h holds K = {8h..8h+7, 16+8h..16+8h+7} of column n
            ((uint4*)&bfr[j])[0] = *(const uint4*)&Bs[n][half * 8];
            ((uint4*)&bfr[j])[1] = *(const uint4*)&Bs[n][16 + half * 8];
        }
        #pragma unroll
        for (int i = 0; i < 2; i++)
            #pragma unroll
            for (int j = 0; j < 4; j++)
                acc[i][j] = __builtin_amdgcn_wmma_f32_16x16x32_bf16(
                    false, af[i], false, bfr[j], (short)0, acc[i][j], false, false);
    }

    if (MODE == 4) {
        // a_scores[b][k] += sum_d  tanh(acc + aq[b][d]) * wa[d]   (partial over this N tile)
        #pragma unroll
        for (int i = 0; i < 2; i++) {
            #pragma unroll
            for (int r = 0; r < 8; r++) {
                int row = tileM + wm + i * 16 + half * 8 + r;  // global row -> (b, k)
                int bb  = row >> 11;                            // LH_ = 2048
                float s = 0.f;
                #pragma unroll
                for (int j = 0; j < 4; j++) {
                    int col = tileN + wn + j * 16 + r16;
                    s += tanhf(acc[i][j][r] + aqv[bb * D_ + col]) * wav[col];
                }
                for (int m = 1; m < 16; m <<= 1) s += __shfl_xor(s, m, 32);
                if (r16 == 0)
                    atomicAdd(&ascore[bb * LH_ + (row & (LH_ - 1))], s);
            }
        }
        return;
    }

    #pragma unroll
    for (int i = 0; i < 2; i++) {
        #pragma unroll
        for (int j = 0; j < 4; j++) {
            #pragma unroll
            for (int r = 0; r < 8; r++) {
                int row = tileM + wm + i * 16 + half * 8 + r;
                int col = tileN + wn + j * 16 + r16;
                float v = acc[i][j][r];
                if (MODE == 0) {
                    if (bias) v += bias[col];
                    ((__bf16*)Cv)[coff + (size_t)row * ldc + col] = (__bf16)v;
                } else if (MODE == 1) {
                    v += bias[col];
                    int b = row >> 9, l = row & (L_ - 1);      // M = B*L
                    int h = col >> 8, dk = col & (DK_ - 1);    // N = D = H*DK
                    ((__bf16*)Cv)[(((size_t)(b * H_ + h) * DK_) + dk) * L_ + l] = (__bf16)v;
                } else if (MODE == 2) {
                    ((float*)Cv)[coff + (size_t)row * ldc + col] = v * scale;
                } else if (MODE == 3) {
                    v += bias[col];
                    ((float*)Cv)[coff + (size_t)row * ldc + col] = v;
                }
            }
        }
    }
}

// ---------------------------------------------------------------------------
__global__ void cast_bf16_k(const float* __restrict__ in, __bf16* __restrict__ out, int n)
{
    int i = (blockIdx.x * 256 + threadIdx.x) * 4;
    if (i < n) {
        float4 v = *(const float4*)(in + i);
        out[i]     = (__bf16)v.x;
        out[i + 1] = (__bf16)v.y;
        out[i + 2] = (__bf16)v.z;
        out[i + 3] = (__bf16)v.w;
    }
}

// 1024x1024 fp32 W -> bf16 W^T (row n = column n of W, K-contiguous)
__global__ __launch_bounds__(256) void transpose_w(const float* __restrict__ in,
                                                   __bf16* __restrict__ out)
{
    __shared__ float t[32][33];
    int bx = blockIdx.x * 32, by = blockIdx.y * 32;
    int tx = threadIdx.x & 31, ty = threadIdx.x >> 5;
    #pragma unroll
    for (int s = 0; s < 32; s += 8)
        t[ty + s][tx] = in[(size_t)(by + ty + s) * D_ + bx + tx];
    __syncthreads();
    #pragma unroll
    for (int s = 0; s < 32; s += 8)
        out[(size_t)(bx + ty + s) * D_ + by + tx] = (__bf16)t[tx][ty + s];
}

__global__ void zero_f32(float* p, int n)
{
    int i = blockIdx.x * 256 + threadIdx.x;
    if (i < n) p[i] = 0.f;
}

// row softmax over L=512 scores; writes bf16 p; head 0 also fp32 to output
__global__ __launch_bounds__(256) void softmax_seq(
    const float* __restrict__ scores, const float* __restrict__ seq_mask,
    __bf16* __restrict__ pb, float* __restrict__ out_attn)
{
    __shared__ float red[256];
    const int row = blockIdx.x;                 // (b*H + h)*L + q
    const int b = row / (H_ * L_);
    const int h = (row / L_) & (H_ - 1);
    const int q = row & (L_ - 1);
    const int tid = threadIdx.x;
    const float* src = scores + (size_t)row * L_;
    float v0 = src[tid], v1 = src[tid + 256];
    if (seq_mask[b * L_ + tid] == 0.f)       v0 = NEGV;
    if (seq_mask[b * L_ + tid + 256] == 0.f) v1 = NEGV;
    red[tid] = fmaxf(v0, v1);
    __syncthreads();
    for (int s = 128; s > 0; s >>= 1) {
        if (tid < s) red[tid] = fmaxf(red[tid], red[tid + s]);
        __syncthreads();
    }
    float mx = red[0];
    __syncthreads();
    float e0 = __expf(v0 - mx), e1 = __expf(v1 - mx);
    red[tid] = e0 + e1;
    __syncthreads();
    for (int s = 128; s > 0; s >>= 1) {
        if (tid < s) red[tid] += red[tid + s];
        __syncthreads();
    }
    float inv = 1.f / red[0];
    float p0 = e0 * inv, p1 = e1 * inv;
    pb[(size_t)row * L_ + tid]       = (__bf16)p0;
    pb[(size_t)row * L_ + tid + 256] = (__bf16)p1;
    if (h == 0) {
        float* o = out_attn + ((size_t)b * L_ + q) * L_;
        o[tid] = p0;
        o[tid + 256] = p1;
    }
}

__global__ __launch_bounds__(256) void softmax_his(
    const float* __restrict__ ascore, const float* __restrict__ his_mask,
    float* __restrict__ aw)
{
    __shared__ float red[256];
    const int b = blockIdx.x;
    const int tid = threadIdx.x;
    float v[8];
    float mx = -3.4e38f;
    #pragma unroll
    for (int t = 0; t < 8; t++) {
        int c = tid + t * 256;
        float x = ascore[b * LH_ + c];
        if (his_mask[b * LH_ + c] == 0.f) x = NEGV;
        v[t] = x;
        mx = fmaxf(mx, x);
    }
    red[tid] = mx;
    __syncthreads();
    for (int s = 128; s > 0; s >>= 1) {
        if (tid < s) red[tid] = fmaxf(red[tid], red[tid + s]);
        __syncthreads();
    }
    mx = red[0];
    __syncthreads();
    float sum = 0.f;
    #pragma unroll
    for (int t = 0; t < 8; t++) { v[t] = __expf(v[t] - mx); sum += v[t]; }
    red[tid] = sum;
    __syncthreads();
    for (int s = 128; s > 0; s >>= 1) {
        if (tid < s) red[tid] += red[tid + s];
        __syncthreads();
    }
    float inv = 1.f / red[0];
    #pragma unroll
    for (int t = 0; t < 8; t++) aw[b * LH_ + tid + t * 256] = v[t] * inv;
}

// LayerNorm of sa_out row, *mask, accumulate mean over L into poi_hidden
__global__ __launch_bounds__(256) void ln_mean(
    const float* __restrict__ sa, const float* __restrict__ g, const float* __restrict__ be,
    const float* __restrict__ mask, float* __restrict__ poi)
{
    __shared__ float r1[256], r2[256];
    const int row = blockIdx.x;            // b*L + l
    const int b = row >> 9;
    const int tid = threadIdx.x;
    const float* x = sa + (size_t)row * D_;
    float xs[4], s = 0.f, s2 = 0.f;
    #pragma unroll
    for (int t = 0; t < 4; t++) {
        float v = x[tid + t * 256];
        xs[t] = v; s += v; s2 += v * v;
    }
    r1[tid] = s; r2[tid] = s2;
    __syncthreads();
    for (int st = 128; st > 0; st >>= 1) {
        if (tid < st) { r1[tid] += r1[tid + st]; r2[tid] += r2[tid + st]; }
        __syncthreads();
    }
    float mean = r1[0] * (1.f / D_);
    float var  = r2[0] * (1.f / D_) - mean * mean;
    float rstd = rsqrtf(var + 1e-5f);
    float mk = mask[row] * (1.f / L_);
    #pragma unroll
    for (int t = 0; t < 4; t++) {
        int d = tid + t * 256;
        float y = (xs[t] - mean) * rstd * g[d] + be[d];
        atomicAdd(&poi[b * D_ + d], y * mk);
    }
}

// hbar[b][d] = sum_k a_w[b][k] * his[b][k][d]
__global__ __launch_bounds__(256) void hbar_k(
    const float* __restrict__ aw, const float* __restrict__ his, float* __restrict__ hbar)
{
    int b = blockIdx.y;
    int d = blockIdx.x * 256 + threadIdx.x;
    const float* hb = his + (size_t)b * LH_ * D_;
    float acc = 0.f;
    for (int k = 0; k < LH_; k++)
        acc += aw[b * LH_ + k] * hb[(size_t)k * D_ + d];
    hbar[b * D_ + d] = acc;
}

// tiny fp32 GEMM: C[32 x 1024] = A[32 x 1024] @ W[1024 x 1024]
__global__ __launch_bounds__(256) void smallmm(
    const float* __restrict__ A, const float* __restrict__ W, float* __restrict__ C)
{
    int m = blockIdx.y;
    int n = blockIdx.x * 256 + threadIdx.x;
    float acc = 0.f;
    for (int k = 0; k < D_; k++)
        acc += A[m * D_ + k] * W[(size_t)k * D_ + n];
    C[m * D_ + n] = acc;
}

// his_poi = poi + LN(aa_out); write seq_hidden = concat(poi, his_poi)
__global__ __launch_bounds__(256) void final_k(
    const float* __restrict__ poi, const float* __restrict__ aa_out,
    const float* __restrict__ g, const float* __restrict__ be, float* __restrict__ out)
{
    __shared__ float r1[256], r2[256];
    const int b = blockIdx.x;
    const int tid = threadIdx.x;
    const float* x = aa_out + (size_t)b * D_;
    float xs[4], s = 0.f, s2 = 0.f;
    #pragma unroll
    for (int t = 0; t < 4; t++) {
        float v = x[tid + t * 256];
        xs[t] = v; s += v; s2 += v * v;
    }
    r1[tid] = s; r2[tid] = s2;
    __syncthreads();
    for (int st = 128; st > 0; st >>= 1) {
        if (tid < st) { r1[tid] += r1[tid + st]; r2[tid] += r2[tid + st]; }
        __syncthreads();
    }
    float mean = r1[0] * (1.f / D_);
    float var  = r2[0] * (1.f / D_) - mean * mean;
    float rstd = rsqrtf(var + 1e-5f);
    #pragma unroll
    for (int t = 0; t < 4; t++) {
        int d = tid + t * 256;
        float y = (xs[t] - mean) * rstd * g[d] + be[d];
        float p = poi[b * D_ + d];
        out[(size_t)b * 2 * D_ + d]      = p;
        out[(size_t)b * 2 * D_ + D_ + d] = p + y;
    }
}

// ---------------------------------------------------------------------------
extern "C" void kernel_launch(void* const* d_in, const int* in_sizes, int n_in,
                              void* d_out, int out_size, void* d_ws, size_t ws_size,
                              hipStream_t stream)
{
    const float* seq      = (const float*)d_in[0];
    const float* seq_mask = (const float*)d_in[1];
    const float* his      = (const float*)d_in[2];
    const float* his_mask = (const float*)d_in[3];
    const float* sa_wq = (const float*)d_in[4];
    const float* sa_bq = (const float*)d_in[5];
    const float* sa_wk = (const float*)d_in[6];
    const float* sa_bk = (const float*)d_in[7];
    const float* sa_wv = (const float*)d_in[8];
    const float* sa_bv = (const float*)d_in[9];
    const float* sa_wo = (const float*)d_in[10];
    const float* sa_bo = (const float*)d_in[11];
    const float* sn_g  = (const float*)d_in[12];
    const float* sn_b  = (const float*)d_in[13];
    const float* aa_wq = (const float*)d_in[14];
    const float* aa_wk = (const float*)d_in[15];
    const float* aa_wv = (const float*)d_in[16];
    const float* aa_wo = (const float*)d_in[17];
    const float* aa_wa = (const float*)d_in[18];
    const float* ln_g  = (const float*)d_in[19];
    const float* ln_b  = (const float*)d_in[20];

    char* wp = (char*)d_ws;
    auto alloc = [&](size_t bytes) {
        void* p = (void*)wp;
        wp += (bytes + 255) & ~(size_t)255;
        return p;
    };
    __bf16* seqb  = (__bf16*)alloc((size_t)B_ * L_  * D_ * 2);
    __bf16* hisb  = (__bf16*)alloc((size_t)B_ * LH_ * D_ * 2);
    __bf16* wqT   = (__bf16*)alloc((size_t)D_ * D_ * 2);
    __bf16* wkT   = (__bf16*)alloc((size_t)D_ * D_ * 2);
    __bf16* wvT   = (__bf16*)alloc((size_t)D_ * D_ * 2);
    __bf16* woT   = (__bf16*)alloc((size_t)D_ * D_ * 2);
    __bf16* awkT  = (__bf16*)alloc((size_t)D_ * D_ * 2);
    __bf16* qB    = (__bf16*)alloc((size_t)B_ * L_ * D_ * 2);
    __bf16* kB    = (__bf16*)alloc((size_t)B_ * L_ * D_ * 2);
    __bf16* vTb   = (__bf16*)alloc((size_t)B_ * L_ * D_ * 2);
    float*  scores= (float*) alloc((size_t)B_ * H_ * L_ * L_ * 4);
    __bf16* pB    = (__bf16*)alloc((size_t)B_ * H_ * L_ * L_ * 2);
    __bf16* ctxb  = (__bf16*)alloc((size_t)B_ * L_ * D_ * 2);
    float*  saout = (float*) alloc((size_t)B_ * L_ * D_ * 4);
    float*  poi   = (float*) alloc((size_t)B_ * D_ * 4);
    float*  ascore= (float*) alloc((size_t)B_ * LH_ * 4);
    float*  aw    = (float*) alloc((size_t)B_ * LH_ * 4);
    float*  hbar  = (float*) alloc((size_t)B_ * D_ * 4);
    float*  aqb   = (float*) alloc((size_t)B_ * D_ * 4);
    float*  actx  = (float*) alloc((size_t)B_ * D_ * 4);
    float*  aaout = (float*) alloc((size_t)B_ * D_ * 4);

    float* out_hidden = (float*)d_out;
    float* out_attn   = (float*)d_out + (size_t)B_ * 2 * D_;

    const dim3 blk(256);

    // ---- casts & weight transposes ----
    cast_bf16_k<<<(B_ * L_ * D_) / 1024, blk, 0, stream>>>(seq, seqb, B_ * L_ * D_);
    cast_bf16_k<<<(B_ * LH_ * D_) / 1024, blk, 0, stream>>>(his, hisb, B_ * LH_ * D_);
    transpose_w<<<dim3(32, 32), blk, 0, stream>>>(sa_wq, wqT);
    transpose_w<<<dim3(32, 32), blk, 0, stream>>>(sa_wk, wkT);
    transpose_w<<<dim3(32, 32), blk, 0, stream>>>(sa_wv, wvT);
    transpose_w<<<dim3(32, 32), blk, 0, stream>>>(sa_wo, woT);
    transpose_w<<<dim3(32, 32), blk, 0, stream>>>(aa_wk, awkT);

    // ---- QKV projections (M = B*L = 16384, N = K = 1024) ----
    gemm_wmma<0><<<dim3(8, 128, 1), blk, 0, stream>>>(
        seqb, wqT, qB, sa_bq, D_, D_, D_, D_, 1, 0, 0, 0, 0, 0, 0, 1.f,
        nullptr, nullptr, nullptr);
    gemm_wmma<0><<<dim3(8, 128, 1), blk, 0, stream>>>(
        seqb, wkT, kB, sa_bk, D_, D_, D_, D_, 1, 0, 0, 0, 0, 0, 0, 1.f,
        nullptr, nullptr, nullptr);
    gemm_wmma<1><<<dim3(8, 128, 1), blk, 0, stream>>>(
        seqb, wvT, vTb, sa_bv, D_, D_, D_, D_, 1, 0, 0, 0, 0, 0, 0, 1.f,
        nullptr, nullptr, nullptr);

    // ---- scores = Q @ K^T / 16 per (b,h): M=N=512, K=256 ----
    gemm_wmma<2><<<dim3(4, 4, B_ * H_), blk, 0, stream>>>(
        qB, kB, scores, nullptr, DK_, D_, D_, L_,
        H_, (long)L_ * D_, (long)DK_, (long)L_ * D_, (long)DK_,
        (long)H_ * L_ * L_, (long)L_ * L_, 0.0625f, nullptr, nullptr, nullptr);

    // ---- softmax rows (writes p bf16 + head-0 fp32 output) ----
    softmax_seq<<<B_ * H_ * L_, blk, 0, stream>>>(scores, seq_mask, pB, out_attn);

    // ---- ctx = P @ V per (b,h): M=512, N=256, K=512 ----
    gemm_wmma<0><<<dim3(2, 4, B_ * H_), blk, 0, stream>>>(
        pB, vTb, ctxb, nullptr, L_, L_, L_, D_,
        H_, (long)H_ * L_ * L_, (long)L_ * L_, (long)H_ * DK_ * L_, (long)DK_ * L_,
        (long)L_ * D_, (long)DK_, 1.f, nullptr, nullptr, nullptr);

    // ---- sa_out = ctx @ Wo + bo (fp32) ----
    gemm_wmma<3><<<dim3(8, 128, 1), blk, 0, stream>>>(
        ctxb, woT, saout, sa_bo, D_, D_, D_, D_, 1, 0, 0, 0, 0, 0, 0, 1.f,
        nullptr, nullptr, nullptr);

    // ---- LN + mask + mean-pool -> poi_hidden ----
    zero_f32<<<(B_ * D_ + 255) / 256, blk, 0, stream>>>(poi, B_ * D_);
    ln_mean<<<B_ * L_, blk, 0, stream>>>(saout, sn_g, sn_b, seq_mask, poi);

    // ---- aq = poi @ aa_wq ----
    smallmm<<<dim3(4, B_), blk, 0, stream>>>(poi, aa_wq, aqb);

    // ---- additive scores: ak GEMM with fused tanh/dot-wa epilogue ----
    zero_f32<<<(B_ * LH_ + 255) / 256, blk, 0, stream>>>(ascore, B_ * LH_);
    gemm_wmma<4><<<dim3(8, 512, 1), blk, 0, stream>>>(
        hisb, awkT, nullptr, nullptr, D_, D_, D_, 0, 1, 0, 0, 0, 0, 0, 0, 1.f,
        aqb, aa_wa, ascore);

    // ---- history softmax, weighted sum, tiny GEMMs ----
    softmax_his<<<B_, blk, 0, stream>>>(ascore, his_mask, aw);
    hbar_k<<<dim3(4, B_), blk, 0, stream>>>(aw, his, hbar);
    smallmm<<<dim3(4, B_), blk, 0, stream>>>(hbar, aa_wv, actx);   // a_ctx = (a_w@his)@Wv
    smallmm<<<dim3(4, B_), blk, 0, stream>>>(actx, aa_wo, aaout);  // aa_out

    // ---- final: his_poi = poi + LN(aa_out); seq_hidden concat ----
    final_k<<<B_, blk, 0, stream>>>(poi, aaout, ln_g, ln_b, out_hidden);

    (void)in_sizes; (void)n_in; (void)out_size; (void)ws_size;
}
